// ChiPredictionLayer_30073361006814
// MI455X (gfx1250) — compile-verified, matrix operand order checked by reading, not run on hardware
//
#include <hip/hip_runtime.h>

// ---------------- problem constants (match reference) ----------------
constexpr int N    = 10000;
constexpr int E    = 160000;
constexpr int D    = 128;
constexpr int H    = 4;
constexpr int ED   = 64;
constexpr int CHI  = 288;
constexpr int PREV = 72;
constexpr int OUTD = 72;
constexpr int C    = 2 * D + CHI + PREV + ED;   // 680
constexpr int CP   = 704;                       // 680 padded to 22*32
constexpr int KC_X = CP / 32;                   // 22 K-chunks for edge GEMM
constexpr int NC_X = H * D + D;                 // 640 fused output cols (4 heads + n1)

// ---------------- types for WMMA ----------------
typedef __attribute__((ext_vector_type(16))) __bf16          v16bf;
typedef __attribute__((ext_vector_type(8)))  float           v8f;
typedef __attribute__((ext_vector_type(8)))  unsigned short  us8;
struct alignas(32) US16 { us8 lo, hi; };

// ---------------- helpers ----------------
__device__ __forceinline__ unsigned short f2bf(float f) {
    unsigned u = __float_as_uint(f);
    unsigned r = u + 0x7FFFu + ((u >> 16) & 1u);   // round-to-nearest-even
    return (unsigned short)(r >> 16);
}
__device__ __forceinline__ float bf2f(unsigned short h) {
    return __uint_as_float(((unsigned)h) << 16);
}
__device__ __forceinline__ float gelu_exact(float x) {
    return 0.5f * x * (1.0f + erff(x * 0.70710678118654752440f));
}
// monotonic float<->uint encoding (enables atomicMax on u32 for segment-max)
__device__ __forceinline__ unsigned encf(float f) {
    unsigned u = __float_as_uint(f);
    return (u & 0x80000000u) ? ~u : (u | 0x80000000u);
}
__device__ __forceinline__ float decf(unsigned u) {
    return __uint_as_float((u & 0x80000000u) ? (u & 0x7FFFFFFFu) : ~u);
}
// Load one 16-element bf16 WMMA fragment. `p` points at 32 contiguous K values
// for this lane's row (A) / column (B); `half` = lane>=16.
// ISA 7.12.2: lo-lanes hold K {0..7, 16..23}, hi-lanes K {8..15, 24..31}.
__device__ __forceinline__ v16bf load_frag(const unsigned short* p, int half) {
    us8 lo = *(const us8*)(p + half * 8);
    us8 hi = *(const us8*)(p + 16 + half * 8);
    US16 t{lo, hi};
    return __builtin_bit_cast(v16bf, t);
}

// ============================================================
// K0: repack fp32 weights into bf16 [kchunk][col][32] fragments
// ============================================================
constexpr int SZ_WCAT  = KC_X * NC_X * 32;  // 450560
constexpr int SZ_WN2   = 4  * 128 * 32;     // 16384
constexpr int SZ_WN3   = 4  * 128 * 32;
constexpr int SZ_WAGGR = 16 * 128 * 32;     // 65536
constexpr int SZ_WD1   = 4  * 512 * 32;     // 65536
constexpr int SZ_WD2   = 16 * 128 * 32;     // 65536
constexpr int SZ_PREP  = SZ_WCAT + SZ_WN2 + SZ_WN3 + SZ_WAGGR + SZ_WD1 + SZ_WD2;

__global__ __launch_bounds__(256) void k_prep(
    const float* __restrict__ aW_w, const float* __restrict__ n1_w,
    const float* __restrict__ n2_w, const float* __restrict__ n3_w,
    const float* __restrict__ aggr_w, const float* __restrict__ d1_w,
    const float* __restrict__ d2_w,
    unsigned short* __restrict__ wcat, unsigned short* __restrict__ wn2,
    unsigned short* __restrict__ wn3, unsigned short* __restrict__ waggr,
    unsigned short* __restrict__ wd1, unsigned short* __restrict__ wd2)
{
    int i = blockIdx.x * 256 + threadIdx.x;
    if (i >= SZ_PREP) return;
    if (i < SZ_WCAT) {
        int kc = i / (NC_X * 32), rem = i % (NC_X * 32);
        int n = rem / 32, kk = rem % 32, K = kc * 32 + kk;
        float v = 0.f;
        if (K < C) {
            if (n < H * D) { int h = n >> 7, d = n & 127; v = aW_w[((size_t)h * C + K) * D + d]; }
            else           { v = n1_w[(size_t)K * D + (n - H * D)]; }
        }
        wcat[i] = f2bf(v); return;
    }
    i -= SZ_WCAT;
    if (i < SZ_WN2) {
        int kc = i / (128 * 32), n = (i / 32) % 128, kk = i % 32;
        wn2[i] = f2bf(n2_w[(size_t)(kc * 32 + kk) * D + n]); return;
    }
    i -= SZ_WN2;
    if (i < SZ_WN3) {
        int kc = i / (128 * 32), n = (i / 32) % 128, kk = i % 32;
        wn3[i] = f2bf(n3_w[(size_t)(kc * 32 + kk) * D + n]); return;
    }
    i -= SZ_WN3;
    if (i < SZ_WAGGR) {
        int kc = i / (128 * 32), n = (i / 32) % 128, kk = i % 32;
        waggr[i] = f2bf(aggr_w[(size_t)(kc * 32 + kk) * D + n]); return;
    }
    i -= SZ_WAGGR;
    if (i < SZ_WD1) {
        int kc = i / (512 * 32), n = (i / 32) % 512, kk = i % 32;
        wd1[i] = f2bf(d1_w[(size_t)(kc * 32 + kk) * (4 * D) + n]); return;
    }
    i -= SZ_WD1;
    {
        int kc = i / (128 * 32), n = (i / 32) % 128, kk = i % 32;
        wd2[i] = f2bf(d2_w[(size_t)(kc * 32 + kk) * D + n]);
    }
}

// ============================================================
// K_init: zero accumulators, smax = encoded(-FLT_MAX)
// ============================================================
__global__ __launch_bounds__(256) void k_init(
    float* __restrict__ agg, float* __restrict__ norm, unsigned* __restrict__ smax)
{
    int i = blockIdx.x * 256 + threadIdx.x;
    if (i < N * H * D) agg[i] = 0.f;
    if (i < N * H) { norm[i] = 0.f; smax[i] = encf(-3.4028234663852886e38f); }
}

// ============================================================
// K1: per-edge fused GEMMs (attention heads + u MLP) + segment max
// 16 edges / block, 256 threads = 8 waves
// ============================================================
__global__ __launch_bounds__(256) void k_edge(
    const float* __restrict__ bb, const float* __restrict__ pc,
    const float* __restrict__ ea, const float* __restrict__ chi,
    const long long* __restrict__ eidx,
    const float* __restrict__ aW_b, const float* __restrict__ aA_w,
    const float* __restrict__ aA_b, const float* __restrict__ n1_b,
    const float* __restrict__ n2_b, const float* __restrict__ n3_b,
    const unsigned short* __restrict__ wcat, const unsigned short* __restrict__ wn2,
    const unsigned short* __restrict__ wn3,
    float* __restrict__ g_scores, unsigned short* __restrict__ g_u,
    unsigned* __restrict__ d_smax)
{
    __shared__ unsigned short sx[16 * CP];        // bf16 A tile (x, gathered)
    __shared__ float          sht[16 * 512];      // post-leaky attention acts
    // reused region: gather phase = f32 async staging of edge_attrs (4KB);
    // after second barrier     = st1 (4KB bf16) + st2 (4KB bf16)
    __shared__ char           sreuse[8192] __attribute__((aligned(16)));
    __shared__ int ssrc[16], ssnk[16];

    float*          stage = (float*)sreuse;                     // [16*64] f32
    unsigned short* st1   = (unsigned short*)sreuse;            // [16*128] bf16
    unsigned short* st2   = (unsigned short*)(sreuse + 4096);   // [16*128] bf16

    const int tid  = threadIdx.x;
    const int lane = tid & 31, wv = tid >> 5;
    const int half = lane >> 4, mrow = lane & 15;
    const int e0   = blockIdx.x * 16;

    // ---- async DMA: contiguous edge_attrs tile (16 x 64 f32 = 4KB) -> LDS ----
    // one b128 per thread; tracked with ASYNCcnt, overlaps the indexed gather
    {
        unsigned long long gaddr =
            (unsigned long long)(ea + (size_t)e0 * ED) + (size_t)tid * 16;
        unsigned ldsoff = (unsigned)(unsigned long long)(void*)(&stage[tid * 4]);
        asm volatile("global_load_async_to_lds_b128 %0, %1, off"
                     :: "v"(ldsoff), "v"(gaddr) : "memory");
    }

    if (tid < 16) {
        ssrc[tid] = (int)eidx[e0 + tid];
        ssnk[tid] = (int)eidx[(size_t)E + e0 + tid];
    }
    asm volatile("s_wait_asynccnt 0" ::: "memory");
    __syncthreads();

    // gather + concat + bf16 convert into LDS (zero-pad C..CP)
    for (int i = tid; i < 16 * CP; i += 256) {
        int e = i / CP, c = i - e * CP;
        float v = 0.f;
        if (c < C) {
            int s = ssrc[e];
            if (c < D)                     v = bb[(size_t)s * D + c];
            else if (c < 2 * D)            v = bb[(size_t)ssnk[e] * D + (c - D)];
            else if (c < 2 * D + CHI)      v = chi[(size_t)s * CHI + (c - 2 * D)];
            else if (c < 2 * D + CHI + ED) v = stage[e * ED + (c - 2 * D - CHI)];
            else                           v = pc[(size_t)s * PREV + (c - 2 * D - CHI - ED)];
        }
        sx[i] = f2bf(v);
    }
    __syncthreads();

    // ---- GEMM1: [16 x 680] @ [680 x 640]  (40 N-tiles, 5 per wave) ----
    v8f acc[5] = {};
    for (int kc = 0; kc < KC_X; ++kc) {
        v16bf a = load_frag(&sx[mrow * CP + kc * 32], half);
        for (int j = 0; j < 5; ++j) {
            int col = (wv + 8 * j) * 16 + mrow;
            const unsigned short* bp = wcat + ((size_t)kc * NC_X + col) * 32;
            if (kc + 1 < KC_X)
                __builtin_prefetch(wcat + ((size_t)(kc + 1) * NC_X + col) * 32, 0, 1);
            v16bf b = load_frag(bp, half);
            acc[j] = __builtin_amdgcn_wmma_f32_16x16x32_bf16(
                false, a, false, b, (short)0, acc[j], false, false);
        }
    }
    // epilogue: bias + leaky (attn) / gelu (n1 stage)
    for (int j = 0; j < 5; ++j) {
        int ncol = (wv + 8 * j) * 16 + mrow;
        for (int i = 0; i < 8; ++i) {
            int m = i + 8 * half;
            float v = acc[j][i];
            if (ncol < H * D) {
                int h = ncol >> 7, d = ncol & 127;
                v += aW_b[h * D + d];
                sht[m * 512 + ncol] = (v > 0.f) ? v : 0.2f * v;
            } else {
                int d = ncol - H * D;
                v += n1_b[d];
                st1[m * 128 + d] = f2bf(gelu_exact(v));
            }
        }
    }
    __syncthreads();

    // ---- attention scores (all 256 threads: 4 lanes per (e,h)) + segment max ----
    {
        int sub = tid & 3;
        int h   = (tid >> 2) & 3;
        int e   = tid >> 4;
        float s = 0.f;
        for (int k = 0; k < 32; ++k) {
            int d = sub * 32 + k;
            s += sht[e * 512 + h * D + d] * aA_w[h * D + d];
        }
        s += __shfl_xor(s, 1, 32);     // butterflies stay inside the quad (wave32)
        s += __shfl_xor(s, 2, 32);
        if (sub == 0) {
            s += aA_b[h];
            g_scores[(size_t)(e0 + e) * H + h] = s;
            atomicMax(&d_smax[(size_t)ssnk[e] * H + h], encf(s));
        }
    }

    // ---- GEMM2: t1 @ n2 (K=128 -> 4 chunks), 8 tiles, 1 per wave ----
    {
        v8f u2 = {};
        int col = wv * 16 + mrow;
        for (int kc = 0; kc < 4; ++kc) {
            v16bf a = load_frag(&st1[mrow * 128 + kc * 32], half);
            v16bf b = load_frag(wn2 + ((size_t)kc * 128 + col) * 32, half);
            u2 = __builtin_amdgcn_wmma_f32_16x16x32_bf16(
                false, a, false, b, (short)0, u2, false, false);
        }
        for (int i = 0; i < 8; ++i) {
            int m = i + 8 * half;
            st2[m * 128 + col] = f2bf(gelu_exact(u2[i] + n2_b[col]));
        }
    }
    __syncthreads();

    // ---- GEMM3: t2 @ n3 -> u, store bf16 ----
    {
        v8f u3 = {};
        int col = wv * 16 + mrow;
        for (int kc = 0; kc < 4; ++kc) {
            v16bf a = load_frag(&st2[mrow * 128 + kc * 32], half);
            v16bf b = load_frag(wn3 + ((size_t)kc * 128 + col) * 32, half);
            u3 = __builtin_amdgcn_wmma_f32_16x16x32_bf16(
                false, a, false, b, (short)0, u3, false, false);
        }
        for (int i = 0; i < 8; ++i) {
            int m = i + 8 * half;
            g_u[(size_t)(e0 + m) * D + col] = f2bf(u3[i] + n3_b[col]);
        }
    }
}

// ============================================================
// K2: softmax numerator + scatter-add into agg / norm
// 4 edges / block, 64 threads per edge
// ============================================================
__global__ __launch_bounds__(256) void k_scatter(
    const float* __restrict__ g_scores, const unsigned* __restrict__ d_smax,
    const unsigned short* __restrict__ g_u, const long long* __restrict__ eidx,
    float* __restrict__ norm, float* __restrict__ agg)
{
    int tid = threadIdx.x;
    int e = blockIdx.x * 4 + (tid >> 6);
    int t = tid & 63;
    int snk = (int)eidx[(size_t)E + e];
    float a[H];
    for (int h = 0; h < H; ++h)
        a[h] = expf(g_scores[(size_t)e * H + h] - decf(d_smax[(size_t)snk * H + h]));
    if (t < H) atomicAdd(&norm[(size_t)snk * H + t], a[t] + 1e-12f);
    float u0 = bf2f(g_u[(size_t)e * D + t]);
    float u1 = bf2f(g_u[(size_t)e * D + t + 64]);
    for (int h = 0; h < H; ++h) {
        atomicAdd(&agg[(size_t)snk * (H * D) + h * D + t],      a[h] * u0);
        atomicAdd(&agg[(size_t)snk * (H * D) + h * D + t + 64], a[h] * u1);
    }
}

// ============================================================
// K3: node update (aggr GEMM, LN, MLP GEMMs, LN, output head)
// 16 nodes / block, 8 waves
// ============================================================
__global__ __launch_bounds__(256) void k_node(
    const float* __restrict__ bb, const float* __restrict__ pc,
    const float* __restrict__ agg, const float* __restrict__ norm,
    const unsigned short* __restrict__ waggr, const unsigned short* __restrict__ wd1,
    const unsigned short* __restrict__ wd2,
    const float* __restrict__ aggr_b, const float* __restrict__ d1_b,
    const float* __restrict__ d2_b, const float* __restrict__ ln_g,
    const float* __restrict__ ln_b, const float* __restrict__ out_w,
    const float* __restrict__ out_b, float* __restrict__ out)
{
    __shared__ unsigned short sa [16 * 512];   // bf16 normalized agg
    __shared__ float          snu[16 * 128];   // nu
    __shared__ unsigned short sx1[16 * 128];   // bf16 LN1 result
    __shared__ unsigned short sth[16 * 512];   // bf16 gelu(x1@d1+b)
    __shared__ float          sx2[16 * 128];
    __shared__ float          sx3[16 * 128];

    const int tid  = threadIdx.x;
    const int lane = tid & 31, wv = tid >> 5;
    const int half = lane >> 4, mrow = lane & 15;
    const int n0   = blockIdx.x * 16;

    for (int i = tid; i < 16 * 512; i += 256) {
        int e = i >> 9, c = i & 511;
        float nm = norm[(size_t)(n0 + e) * H + (c >> 7)];
        float v  = (nm > 0.f) ? agg[(size_t)(n0 + e) * 512 + c] / nm : 0.f;
        sa[i] = f2bf(v);
    }
    __syncthreads();

    // nu = anorm @ aggr_w + b   (K=512 -> 16 chunks, 8 tiles)
    {
        v8f acc = {};
        int col = wv * 16 + mrow;
        for (int kc = 0; kc < 16; ++kc) {
            v16bf a = load_frag(&sa[mrow * 512 + kc * 32], half);
            v16bf b = load_frag(waggr + ((size_t)kc * 128 + col) * 32, half);
            acc = __builtin_amdgcn_wmma_f32_16x16x32_bf16(
                false, a, false, b, (short)0, acc, false, false);
        }
        for (int i = 0; i < 8; ++i)
            snu[(i + 8 * half) * 128 + col] = acc[i] + aggr_b[col];
    }
    __syncthreads();

    // LN1 over (bb + nu): 16 lanes per node, butterfly within 16-lane half-wave
    {
        int e = tid >> 4, sub = tid & 15, node = n0 + e;
        float tv[8];
        float s = 0.f, sq = 0.f;
        for (int k = 0; k < 8; ++k) {
            int d = sub * 8 + k;
            float t = bb[(size_t)node * D + d] + snu[e * 128 + d];
            tv[k] = t; s += t; sq += t * t;
        }
        for (int m = 1; m < 16; m <<= 1) {
            s  += __shfl_xor(s,  m, 32);
            sq += __shfl_xor(sq, m, 32);
        }
        float mu = s * (1.f / 128.f), var = sq * (1.f / 128.f) - mu * mu;
        float rs = rsqrtf(var + 1e-5f);
        for (int k = 0; k < 8; ++k) {
            int d = sub * 8 + k;
            sx1[e * 128 + d] = f2bf((tv[k] - mu) * rs * ln_g[d] + ln_b[d]);
        }
    }
    __syncthreads();

    // x1 @ d1 (128->512), gelu   (32 tiles, 4 per wave)
    {
        v8f acc[4] = {};
        for (int kc = 0; kc < 4; ++kc) {
            v16bf a = load_frag(&sx1[mrow * 128 + kc * 32], half);
            for (int j = 0; j < 4; ++j) {
                int col = (wv + 8 * j) * 16 + mrow;
                v16bf b = load_frag(wd1 + ((size_t)kc * 512 + col) * 32, half);
                acc[j] = __builtin_amdgcn_wmma_f32_16x16x32_bf16(
                    false, a, false, b, (short)0, acc[j], false, false);
            }
        }
        for (int j = 0; j < 4; ++j) {
            int col = (wv + 8 * j) * 16 + mrow;
            for (int i = 0; i < 8; ++i)
                sth[(i + 8 * half) * 512 + col] = f2bf(gelu_exact(acc[j][i] + d1_b[col]));
        }
    }
    __syncthreads();

    // (.) @ d2 (512->128)
    {
        v8f acc = {};
        int col = wv * 16 + mrow;
        for (int kc = 0; kc < 16; ++kc) {
            v16bf a = load_frag(&sth[mrow * 512 + kc * 32], half);
            v16bf b = load_frag(wd2 + ((size_t)kc * 128 + col) * 32, half);
            acc = __builtin_amdgcn_wmma_f32_16x16x32_bf16(
                false, a, false, b, (short)0, acc, false, false);
        }
        for (int i = 0; i < 8; ++i)
            sx2[(i + 8 * half) * 128 + col] = acc[i] + d2_b[col];
    }
    __syncthreads();

    // LN2 over (x2 + nu): same 16-lane-per-node scheme
    {
        int e = tid >> 4, sub = tid & 15;
        float tv[8];
        float s = 0.f, sq = 0.f;
        for (int k = 0; k < 8; ++k) {
            int d = sub * 8 + k;
            float t = sx2[e * 128 + d] + snu[e * 128 + d];
            tv[k] = t; s += t; sq += t * t;
        }
        for (int m = 1; m < 16; m <<= 1) {
            s  += __shfl_xor(s,  m, 32);
            sq += __shfl_xor(sq, m, 32);
        }
        float mu = s * (1.f / 128.f), var = sq * (1.f / 128.f) - mu * mu;
        float rs = rsqrtf(var + 1e-5f);
        for (int k = 0; k < 8; ++k) {
            int d = sub * 8 + k;
            sx3[e * 128 + d] = (tv[k] - mu) * rs * ln_g[d] + ln_b[d];
        }
    }
    __syncthreads();

    // output head + prev_chi copy
    for (int p = tid; p < 16 * OUTD; p += 256) {
        int e = p / OUTD, o = p % OUTD;
        float s = out_b[o];
        for (int d = 0; d < D; ++d) s += sx3[e * 128 + d] * out_w[(size_t)d * OUTD + o];
        out[(size_t)(n0 + e) * (PREV + OUTD) + PREV + o] = s;
    }
    for (int p = tid; p < 16 * PREV; p += 256) {
        int e = p / PREV, o = p % PREV;
        out[(size_t)(n0 + e) * (PREV + OUTD) + o] = pc[(size_t)(n0 + e) * PREV + o];
    }
}

// ============================================================
// launch
// ============================================================
extern "C" void kernel_launch(void* const* d_in, const int* in_sizes, int n_in,
                              void* d_out, int out_size, void* d_ws, size_t ws_size,
                              hipStream_t stream) {
    const float*     bb   = (const float*)d_in[0];
    const float*     pc   = (const float*)d_in[1];
    const float*     ea   = (const float*)d_in[2];
    const float*     chi  = (const float*)d_in[3];
    const long long* eidx = (const long long*)d_in[4];
    const float* aW_w = (const float*)d_in[5];
    const float* aW_b = (const float*)d_in[6];
    const float* aA_w = (const float*)d_in[7];
    const float* aA_b = (const float*)d_in[8];
    const float* n1_w = (const float*)d_in[9];
    const float* n1_b = (const float*)d_in[10];
    const float* n2_w = (const float*)d_in[11];
    const float* n2_b = (const float*)d_in[12];
    const float* n3_w = (const float*)d_in[13];
    const float* n3_b = (const float*)d_in[14];
    const float* d1_w = (const float*)d_in[15];
    const float* d1_b = (const float*)d_in[16];
    const float* d2_w = (const float*)d_in[17];
    const float* d2_b = (const float*)d_in[18];
    const float* aggr_w = (const float*)d_in[19];
    const float* aggr_b = (const float*)d_in[20];
    const float* ln_g = (const float*)d_in[21];
    const float* ln_b = (const float*)d_in[22];
    const float* out_w = (const float*)d_in[23];
    const float* out_b = (const float*)d_in[24];
    float* out = (float*)d_out;

    // ---- workspace carve-up ----
    char* base = (char*)d_ws;
    size_t off = 0;
    auto carve = [&](size_t bytes) { char* p = base + off; off = (off + bytes + 255) & ~(size_t)255; return p; };
    unsigned*       g_smax   = (unsigned*)      carve((size_t)N * H * 4);
    float*          g_norm   = (float*)         carve((size_t)N * H * 4);
    float*          g_agg    = (float*)         carve((size_t)N * H * D * 4);
    float*          g_scores = (float*)         carve((size_t)E * H * 4);
    unsigned short* g_u      = (unsigned short*)carve((size_t)E * D * 2);
    unsigned short* g_wcat   = (unsigned short*)carve((size_t)SZ_WCAT * 2);
    unsigned short* g_wn2    = (unsigned short*)carve((size_t)SZ_WN2 * 2);
    unsigned short* g_wn3    = (unsigned short*)carve((size_t)SZ_WN3 * 2);
    unsigned short* g_waggr  = (unsigned short*)carve((size_t)SZ_WAGGR * 2);
    unsigned short* g_wd1    = (unsigned short*)carve((size_t)SZ_WD1 * 2);
    unsigned short* g_wd2    = (unsigned short*)carve((size_t)SZ_WD2 * 2);

    k_prep<<<(SZ_PREP + 255) / 256, 256, 0, stream>>>(
        aW_w, n1_w, n2_w, n3_w, aggr_w, d1_w, d2_w,
        g_wcat, g_wn2, g_wn3, g_waggr, g_wd1, g_wd2);

    k_init<<<(N * H * D + 255) / 256, 256, 0, stream>>>(g_agg, g_norm, g_smax);

    k_edge<<<E / 16, 256, 0, stream>>>(
        bb, pc, ea, chi, eidx, aW_b, aA_w, aA_b, n1_b, n2_b, n3_b,
        g_wcat, g_wn2, g_wn3, g_scores, g_u, g_smax);

    k_scatter<<<E / 4, 256, 0, stream>>>(g_scores, g_smax, g_u, eidx, g_norm, g_agg);

    k_node<<<N / 16, 256, 0, stream>>>(
        bb, pc, g_agg, g_norm, g_waggr, g_wd1, g_wd2,
        aggr_b, d1_b, d2_b, ln_g, ln_b, out_w, out_b, out);
}
